// TracGCN_65317862637683
// MI455X (gfx1250) — matrix operand order repaired and checked by reading, not compiled, not run on hardware
//
#include <hip/hip_runtime.h>
#include <hip/hip_bf16.h>

typedef __attribute__((ext_vector_type(16))) _Float16 v16h;
typedef __attribute__((ext_vector_type(8)))  float    v8f;

#define NN_C 262144
#define NE_C 8388608
#define B_C  32
#define THREADS 256

// ---------------- sparse graph kernels ----------------

__global__ void k_deg(const int* __restrict__ src, const int* __restrict__ dst,
                      float* __restrict__ deg, int ne) {
    int e = blockIdx.x * blockDim.x + threadIdx.x;
    if (e >= ne) return;
    int s = src[e];
    if (s != dst[e]) atomicAdd(&deg[s], 1.0f);
}

__global__ void k_dinv(const float* __restrict__ deg, float* __restrict__ dinv, int nn) {
    int n = blockIdx.x * blockDim.x + threadIdx.x;
    if (n >= nn) return;
    float d = deg[n];
    dinv[n] = (d > 0.0f) ? rsqrtf(fmaxf(d, 1.0f)) : 0.0f;
}

__global__ void k_norm(const int* __restrict__ src, const int* __restrict__ dst,
                       const float* __restrict__ dinv, float* __restrict__ nrm, int ne) {
    int e = blockIdx.x * blockDim.x + threadIdx.x;
    if (e >= ne) return;
    int s = src[e], d = dst[e];
    nrm[e] = (s != d) ? -(dinv[s] * dinv[d]) : 0.0f;
}

// y[dst] += norm * t[src]   (1 channel)
__global__ void k_lmv1(const int* __restrict__ src, const int* __restrict__ dst,
                       const float* __restrict__ nrm, const float* __restrict__ t,
                       float* __restrict__ y, int ne) {
    int e = blockIdx.x * blockDim.x + threadIdx.x;
    if (e >= ne) return;
    float s = nrm[e];
    if (s == 0.0f) return;
    atomicAdd(&y[dst[e]], s * t[src[e]]);
}

// y[dst][0..7] += norm * t[src][0..7]   (8 channels)
__global__ void k_lmv8(const int* __restrict__ src, const int* __restrict__ dst,
                       const float* __restrict__ nrm, const float* __restrict__ t,
                       float* __restrict__ y, int ne) {
    int e = blockIdx.x * blockDim.x + threadIdx.x;
    if (e >= ne) return;
    float s = nrm[e];
    if (s == 0.0f) return;
    const float* ts = t + (size_t)src[e] * 8;
    float*       yd = y + (size_t)dst[e] * 8;
#pragma unroll
    for (int c = 0; c < 8; ++c) atomicAdd(&yd[c], s * ts[c]);
}

// next = 2*y - prev  (prev may alias next)
__global__ void k_cheb_next(const float* __restrict__ y, const float* __restrict__ prev,
                            float* __restrict__ next, int len) {
    int i = blockIdx.x * blockDim.x + threadIdx.x;
    if (i >= len) return;
    next[i] = 2.0f * y[i] - prev[i];
}

// out[n][c] (+)= t[n] * Wk[c]   (layer-1 channel expansion, Wk is 8 floats)
__global__ void k_l1_acc(const float* __restrict__ t, const float* __restrict__ Wk,
                         float* __restrict__ out, int nn, int overwrite) {
    int n = blockIdx.x * blockDim.x + threadIdx.x;
    if (n >= nn) return;
    float tv = t[n];
    float* o = out + (size_t)n * 8;
#pragma unroll
    for (int c = 0; c < 8; ++c) {
        float v = tv * Wk[c];
        o[c] = overwrite ? v : (o[c] + v);
    }
}

// out[n][c] (+)= sum_j T[n][j] * Wk[j][c]   (layer-2 8x8 per-node matmul)
__global__ void k_l2_acc(const float* __restrict__ T, const float* __restrict__ Wk,
                         float* __restrict__ out, int nn, int overwrite) {
    int n = blockIdx.x * blockDim.x + threadIdx.x;
    if (n >= nn) return;
    const float* tr = T + (size_t)n * 8;
    float tv[8];
#pragma unroll
    for (int j = 0; j < 8; ++j) tv[j] = tr[j];
    float* o = out + (size_t)n * 8;
#pragma unroll
    for (int c = 0; c < 8; ++c) {
        float v = 0.0f;
#pragma unroll
        for (int j = 0; j < 8; ++j) v += tv[j] * Wk[j * 8 + c];
        o[c] = overwrite ? v : (o[c] + v);
    }
}

// out = act(acc + bias[n])   act: 0=none 1=relu 2=elu ; acc may alias out
__global__ void k_bias_act(const float* __restrict__ acc, const float* __restrict__ bias,
                           float* __restrict__ out, int M, int N, int act) {
    int i = blockIdx.x * blockDim.x + threadIdx.x;
    if (i >= M * N) return;
    float v = acc[i] + bias[i % N];
    if (act == 1)      v = fmaxf(v, 0.0f);
    else if (act == 2) v = (v > 0.0f) ? v : (__expf(v) - 1.0f);
    out[i] = v;
}

// ---------------- WMMA split-K GEMM:  Cacc[M,N] += A[M,K] @ W[K,N] ----------------
// One wave handles a 16x32 output tile (two 16x16 WMMA tiles sharing A) per K-split.
// Branchless loads: OOB columns get clamped addresses (their results are never
// stored); only the K-tail block uses value masking. f32->f16 inputs, f32 acc,
// atomic f32 reduction across K-splits. EXEC stays all-ones for the WMMAs.
__global__ void k_gemm_wmma(const float* __restrict__ A, const float* __restrict__ W,
                            float* __restrict__ Cacc, int M, int K, int N, int k_per_split) {
    const int lane = threadIdx.x;          // 0..31, wave32
    const int col0 = blockIdx.x * 32;      // two 16-col tiles
    const int row0 = blockIdx.y * 16;
    const int k0   = blockIdx.z * k_per_split;
    int k1 = k0 + k_per_split; if (k1 > K) k1 = K;

    const int rA  = lane & 15;             // A row within tile (lane%16)
    const int kgA = (lane >> 4) * 16;      // A K half (matches B layout)
    const int cB  = lane & 15;             // B col within sub-tile
    const int kgB = (lane >> 4) * 16;      // B K half (contiguous k = 16h+i, same as A)

    const float* Arow = A + (size_t)(row0 + rA) * K;
    // clamp OOB columns to a valid address; their products land in D columns we never store
    const float* Wc0 = W + min(col0 + cB,      N - 1);
    const float* Wc1 = W + min(col0 + 16 + cB, N - 1);

    v8f acc0 = {}, acc1 = {};
    int kb = k0;

    // ---- fast path: full 32-deep K blocks, zero predicates ----
    for (; kb + 32 <= k1; kb += 32) {
        if (kb + 64 <= k1)   // prefetch next weight K-slab (global_prefetch_b8)
            __builtin_prefetch(Wc0 + (size_t)(kb + 32 + kgB) * N, 0, 1);
        const float* Ap  = Arow + kb + kgA;
        const float* Wk0 = Wc0 + (size_t)(kb + kgB) * N;
        const float* Wk1 = Wc1 + (size_t)(kb + kgB) * N;
        v16h a, b0, b1;
#pragma unroll
        for (int i = 0; i < 8; ++i) a[i]     = (_Float16)Ap[i];        // K kgA+0..7
#pragma unroll
        for (int i = 0; i < 8; ++i) a[8 + i] = (_Float16)Ap[8 + i];    // K kgA+8..15
#pragma unroll
        for (int i = 0; i < 16; ++i) b0[i] = (_Float16)Wk0[(size_t)i * N];
#pragma unroll
        for (int i = 0; i < 16; ++i) b1[i] = (_Float16)Wk1[(size_t)i * N];
        acc0 = __builtin_amdgcn_wmma_f32_16x16x32_f16(false, a, false, b0,
                                                      (short)0, acc0, false, false);
        acc1 = __builtin_amdgcn_wmma_f32_16x16x32_f16(false, a, false, b1,
                                                      (short)0, acc1, false, false);
    }

    // ---- K tail (k1 - kb in (0,32)): branchless clamped loads + value mask ----
    if (kb < k1) {
        v16h a, b0, b1;
#pragma unroll
        for (int i = 0; i < 16; ++i) {
            int kk = kb + kgA + i;
            float av = Arow[min(kk, K - 1)];
            a[i] = (_Float16)((kk < k1) ? av : 0.0f);
        }
#pragma unroll
        for (int i = 0; i < 16; ++i) {
            int kk = kb + kgB + i;
            int kc = min(kk, K - 1);
            float w0 = Wc0[(size_t)kc * N];
            float w1 = Wc1[(size_t)kc * N];
            bool ok = kk < k1;
            b0[i] = (_Float16)(ok ? w0 : 0.0f);
            b1[i] = (_Float16)(ok ? w1 : 0.0f);
        }
        acc0 = __builtin_amdgcn_wmma_f32_16x16x32_f16(false, a, false, b0,
                                                      (short)0, acc0, false, false);
        acc1 = __builtin_amdgcn_wmma_f32_16x16x32_f16(false, a, false, b1,
                                                      (short)0, acc1, false, false);
    }

    // C/D layout: VGPR i -> row = 8*(lane/16)+i, col = lane%16
    const int rbase = row0 + ((lane >> 4) * 8);
    const int cc0 = col0 + (lane & 15);
    const int cc1 = cc0 + 16;
    if (cc0 < N) {
#pragma unroll
        for (int i = 0; i < 8; ++i) {
            int r = rbase + i;
            if (r < M) atomicAdd(&Cacc[(size_t)r * N + cc0], acc0[i]);
        }
    }
    if (cc1 < N) {
#pragma unroll
        for (int i = 0; i < 8; ++i) {
            int r = rbase + i;
            if (r < M) atomicAdd(&Cacc[(size_t)r * N + cc1], acc1[i]);
        }
    }
}

// final: out[b] = clip(concat(g2,m4)[b] @ Wo + bo, 0, 110)
__global__ void k_final(const float* __restrict__ g2, const float* __restrict__ m4,
                        const float* __restrict__ Wo, const float* __restrict__ bo,
                        float* __restrict__ out) {
    int b = threadIdx.x;
    if (b >= B_C) return;
    float s = bo[0];
    for (int j = 0; j < 50; ++j) s += g2[b * 50 + j] * Wo[j];
    for (int j = 0; j < 50; ++j) s += m4[b * 50 + j] * Wo[50 + j];
    out[b] = fminf(fmaxf(s, 0.0f), 110.0f);
}

// ---------------- host orchestration ----------------

extern "C" void kernel_launch(void* const* d_in, const int* in_sizes, int n_in,
                              void* d_out, int out_size, void* d_ws, size_t ws_size,
                              hipStream_t stream) {
    (void)in_sizes; (void)n_in; (void)out_size; (void)ws_size;

    const float* x   = (const float*)d_in[0];
    const int*   ei  = (const int*)  d_in[1];
    const float* W1  = (const float*)d_in[4];   // [5,1,8]
    const float* b1  = (const float*)d_in[5];
    const float* W2  = (const float*)d_in[6];   // [5,8,8]
    const float* b2  = (const float*)d_in[7];
    const float* Wg1 = (const float*)d_in[8];   // [65536,1000]
    const float* bg1 = (const float*)d_in[9];
    const float* Wg2 = (const float*)d_in[10];  // [1000,50]
    const float* bg2 = (const float*)d_in[11];
    const float* Wm1 = (const float*)d_in[12];  // [8192,350]
    const float* bm1 = (const float*)d_in[13];
    const float* Wm2 = (const float*)d_in[14];  // [350,350]
    const float* bm2 = (const float*)d_in[15];
    const float* Wm3 = (const float*)d_in[16];  // [350,350]
    const float* bm3 = (const float*)d_in[17];
    const float* Wm4 = (const float*)d_in[18];  // [350,50]
    const float* bm4 = (const float*)d_in[19];
    const float* Wo  = (const float*)d_in[20];  // [100,1]
    const float* bo  = (const float*)d_in[21];
    float* out = (float*)d_out;

    const int* src = ei;
    const int* dst = ei + NE_C;

    char* ws = (char*)d_ws;
    size_t off = 0;
    auto alloc = [&](size_t bytes) -> float* {
        float* p = (float*)(ws + off);
        off += (bytes + 255) & ~(size_t)255;
        return p;
    };

    float* deg  = alloc((size_t)NN_C * 4);
    float* dinv = alloc((size_t)NN_C * 4);
    float* nrm  = alloc((size_t)NE_C * 4);
    float* y1   = alloc((size_t)NN_C * 4);
    float* Ta   = alloc((size_t)NN_C * 4);
    float* Tb   = alloc((size_t)NN_C * 4);
    float* h1   = alloc((size_t)NN_C * 8 * 4);
    float* y2   = alloc((size_t)NN_C * 8 * 4);
    float* Ta2  = alloc((size_t)NN_C * 8 * 4);
    float* Tb2  = alloc((size_t)NN_C * 8 * 4);
    float* h2   = alloc((size_t)NN_C * 8 * 4);
    float* acc  = alloc((size_t)B_C * 1000 * 4);   // reused across all GEMMs
    float* g1   = alloc((size_t)B_C * 1000 * 4);
    float* g2   = alloc((size_t)B_C * 50 * 4);
    float* m1   = alloc((size_t)B_C * 350 * 4);
    float* m2   = alloc((size_t)B_C * 350 * 4);
    float* m3   = alloc((size_t)B_C * 350 * 4);
    float* m4   = alloc((size_t)B_C * 50 * 4);

    auto cdiv = [](int a, int b) { return (a + b - 1) / b; };
    const int gNE = cdiv(NE_C, THREADS);
    const int gNN = cdiv(NN_C, THREADS);
    const int gN8 = cdiv(NN_C * 8, THREADS);

    auto gemm = [&](const float* A, const float* Wm, const float* bias, float* o,
                    int M, int K, int N, int splitk, int act) {
        hipMemsetAsync(acc, 0, (size_t)M * N * 4, stream);
        int kps = cdiv(cdiv(K, splitk), 32) * 32;          // K per split, multiple of 32
        dim3 grid(cdiv(N, 32), cdiv(M, 16), splitk);       // 32 cols per wave (2 sub-tiles)
        k_gemm_wmma<<<grid, 32, 0, stream>>>(A, Wm, acc, M, K, N, kps);
        k_bias_act<<<cdiv(M * N, THREADS), THREADS, 0, stream>>>(acc, bias, o, M, N, act);
    };

    // --- graph normalization ---
    hipMemsetAsync(deg, 0, (size_t)NN_C * 4, stream);
    k_deg <<<gNE, THREADS, 0, stream>>>(src, dst, deg, NE_C);
    k_dinv<<<gNN, THREADS, 0, stream>>>(deg, dinv, NN_C);
    k_norm<<<gNE, THREADS, 0, stream>>>(src, dst, dinv, nrm, NE_C);

    // --- ChebConv layer 1 (C: 1 -> 8), Tx0 = x ---
    k_l1_acc<<<gNN, THREADS, 0, stream>>>(x, W1 + 0, h1, NN_C, 1);
    // Tx1 = L x
    hipMemsetAsync(Ta, 0, (size_t)NN_C * 4, stream);
    k_lmv1<<<gNE, THREADS, 0, stream>>>(src, dst, nrm, x, Ta, NE_C);
    k_l1_acc<<<gNN, THREADS, 0, stream>>>(Ta, W1 + 8, h1, NN_C, 0);
    // Tx2 = 2 L Tx1 - Tx0
    hipMemsetAsync(y1, 0, (size_t)NN_C * 4, stream);
    k_lmv1<<<gNE, THREADS, 0, stream>>>(src, dst, nrm, Ta, y1, NE_C);
    k_cheb_next<<<gNN, THREADS, 0, stream>>>(y1, x, Tb, NN_C);
    k_l1_acc<<<gNN, THREADS, 0, stream>>>(Tb, W1 + 16, h1, NN_C, 0);
    // Tx3 = 2 L Tx2 - Tx1  (overwrite Ta)
    hipMemsetAsync(y1, 0, (size_t)NN_C * 4, stream);
    k_lmv1<<<gNE, THREADS, 0, stream>>>(src, dst, nrm, Tb, y1, NE_C);
    k_cheb_next<<<gNN, THREADS, 0, stream>>>(y1, Ta, Ta, NN_C);
    k_l1_acc<<<gNN, THREADS, 0, stream>>>(Ta, W1 + 24, h1, NN_C, 0);
    // Tx4 = 2 L Tx3 - Tx2  (overwrite Tb)
    hipMemsetAsync(y1, 0, (size_t)NN_C * 4, stream);
    k_lmv1<<<gNE, THREADS, 0, stream>>>(src, dst, nrm, Ta, y1, NE_C);
    k_cheb_next<<<gNN, THREADS, 0, stream>>>(y1, Tb, Tb, NN_C);
    k_l1_acc<<<gNN, THREADS, 0, stream>>>(Tb, W1 + 32, h1, NN_C, 0);
    // h1 = relu(h1 + b1)
    k_bias_act<<<gN8, THREADS, 0, stream>>>(h1, b1, h1, NN_C, 8, 1);

    // --- ChebConv layer 2 (C: 8 -> 8), Tx0 = h1 ---
    k_l2_acc<<<gNN, THREADS, 0, stream>>>(h1, W2 + 0, h2, NN_C, 1);
    // Tx1 = L h1
    hipMemsetAsync(Ta2, 0, (size_t)NN_C * 8 * 4, stream);
    k_lmv8<<<gNE, THREADS, 0, stream>>>(src, dst, nrm, h1, Ta2, NE_C);
    k_l2_acc<<<gNN, THREADS, 0, stream>>>(Ta2, W2 + 64, h2, NN_C, 0);
    // Tx2
    hipMemsetAsync(y2, 0, (size_t)NN_C * 8 * 4, stream);
    k_lmv8<<<gNE, THREADS, 0, stream>>>(src, dst, nrm, Ta2, y2, NE_C);
    k_cheb_next<<<gN8, THREADS, 0, stream>>>(y2, h1, Tb2, NN_C * 8);
    k_l2_acc<<<gNN, THREADS, 0, stream>>>(Tb2, W2 + 128, h2, NN_C, 0);
    // Tx3 (overwrite Ta2)
    hipMemsetAsync(y2, 0, (size_t)NN_C * 8 * 4, stream);
    k_lmv8<<<gNE, THREADS, 0, stream>>>(src, dst, nrm, Tb2, y2, NE_C);
    k_cheb_next<<<gN8, THREADS, 0, stream>>>(y2, Ta2, Ta2, NN_C * 8);
    k_l2_acc<<<gNN, THREADS, 0, stream>>>(Ta2, W2 + 192, h2, NN_C, 0);
    // Tx4 (overwrite Tb2)
    hipMemsetAsync(y2, 0, (size_t)NN_C * 8 * 4, stream);
    k_lmv8<<<gNE, THREADS, 0, stream>>>(src, dst, nrm, Ta2, y2, NE_C);
    k_cheb_next<<<gN8, THREADS, 0, stream>>>(y2, Tb2, Tb2, NN_C * 8);
    k_l2_acc<<<gNN, THREADS, 0, stream>>>(Tb2, W2 + 256, h2, NN_C, 0);
    // h2 = relu(h2 + b2)
    k_bias_act<<<gN8, THREADS, 0, stream>>>(h2, b2, h2, NN_C, 8, 1);

    // --- dense heads (all via WMMA split-K GEMM) ---
    // g branch: h2 viewed as [32, 65536]
    gemm(h2, Wg1, bg1, g1, B_C, NN_C / B_C * 8 /*65536*/, 1000, 32, 1);
    gemm(g1, Wg2, bg2, g2, B_C, 1000, 50, 1, 1);
    // m branch: x viewed as [32, 8192]
    gemm(x,  Wm1, bm1, m1, B_C, 8192, 350, 8, 2);
    gemm(m1, Wm2, bm2, m2, B_C, 350, 350, 1, 2);
    gemm(m2, Wm3, bm3, m3, B_C, 350, 350, 1, 2);
    gemm(m3, Wm4, bm4, m4, B_C, 350, 50, 1, 0);

    k_final<<<1, 32, 0, stream>>>(g2, m4, Wo, bo, out);
}
